// TemporalMambaFusionC1_89945205113390
// MI455X (gfx1250) — compile-verified, matrix operand order checked by reading, not run on hardware
//
#include <hip/hip_runtime.h>
#include <math.h>

// ---------------------------------------------------------------------------
// TemporalMambaFusion for MI455X (gfx1250): bf16 WMMA everywhere.
//   tensors: B=2, T=8, C=64, H=W=128, HIDDEN=128, GN groups=8
// ---------------------------------------------------------------------------

#define B_    2
#define T_    8
#define C_    64
#define HID_  128
#define H_    128
#define W_    128
#define NPIX  (B_*H_*W_)   // 32768

typedef __bf16 bf16_t;
typedef __attribute__((ext_vector_type(8)))  __bf16 v8bf;
typedef __attribute__((ext_vector_type(16))) __bf16 v16bf;
typedef __attribute__((ext_vector_type(8)))  float  v8f;

// ----------------------------- scalar helpers ------------------------------

__device__ __forceinline__ bf16_t f2bf(float f) {
  unsigned int u = __builtin_bit_cast(unsigned int, f);
  u += 0x7FFFu + ((u >> 16) & 1u);                      // round-to-nearest-even
  unsigned short s = (unsigned short)(u >> 16);
  return __builtin_bit_cast(bf16_t, s);
}
__device__ __forceinline__ float bf2f(bf16_t b) {
  unsigned short s = __builtin_bit_cast(unsigned short, b);
  unsigned int u = ((unsigned int)s) << 16;
  return __builtin_bit_cast(float, u);
}
__device__ __forceinline__ float gelu_f(float x) {
  return 0.5f * x * (1.0f + erff(x * 0.70710678118654752f));   // exact (approximate=False)
}
__device__ __forceinline__ float sigm_f(float x) { return 1.0f / (1.0f + expf(-x)); }

// ----------------------------- WMMA helpers --------------------------------

__device__ __forceinline__ v8f wmma_bf16(v16bf a, v16bf b, v8f c) {
  // D = A(16x32 bf16) x B(32x16 bf16) + C(16x16 f32)
  return __builtin_amdgcn_wmma_f32_16x16x32_bf16(false, a, false, b, (short)0, c,
                                                 false, false);
}

// contiguous 16 bf16 (two 16B loads -> no 32B alignment requirement)
__device__ __forceinline__ v16bf load16(const bf16_t* p) {
  v8bf lo = *(const v8bf*)p;
  v8bf hi = *(const v8bf*)(p + 8);
  return __builtin_shufflevector(lo, hi, 0,1,2,3,4,5,6,7,8,9,10,11,12,13,14,15);
}

// A-fragment of a row-major 16xK bf16 tile, K-step starting at k0.
// Per ISA 7.12.2 (16-bit A 16x32): lane[0..15] rows, lane>=16 same rows but
// K offset by 8 (k 8-15, 24-31); per lane the 16 values are two contiguous 8-elem runs.
__device__ __forceinline__ v16bf load_a(const bf16_t* base, int stride, int k0, int lane) {
  int row = lane & 15;
  int khalf = (lane >> 4) & 1;
  const bf16_t* p = base + row * stride + k0 + khalf * 8;
  v8bf lo = *(const v8bf*)p;
  v8bf hi = *(const v8bf*)(p + 16);
  return __builtin_shufflevector(lo, hi, 0,1,2,3,4,5,6,7,8,9,10,11,12,13,14,15);
}

#define WAVE_SYNC() asm volatile("s_wait_dscnt 0" ::: "memory")

// ---------------------------------------------------------------------------
// Weight swizzle: pack B-matrix [K][N] fp32 into per-lane WMMA fragments:
//   dst[kt][nt][lane][16] bf16, so a B-frag is one contiguous 32B read.
// ---------------------------------------------------------------------------

__global__ void __launch_bounds__(256)
swz_linear_kernel(const float* __restrict__ src, bf16_t* __restrict__ dst, int K, int N) {
  int idx = blockIdx.x * 256 + threadIdx.x;
  if (idx >= K * N) return;
  int NT = N >> 4;
  int j = idx & 15, lane = (idx >> 4) & 31, q = idx >> 9;
  int nt = q % NT, kt = q / NT;
  int n = nt * 16 + (lane & 15);
  int khalf = lane >> 4;
  int k = kt * 32 + ((j < 8) ? (khalf * 8 + j) : (16 + khalf * 8 + (j - 8)));
  dst[idx] = f2bf(src[k * N + n]);
}

// conv weights OIHW (64,64,3,3) -> dst[tap][kt][nt][lane][16], k=ci, n=co
__global__ void __launch_bounds__(256)
swz_conv_kernel(const float* __restrict__ src, bf16_t* __restrict__ dst) {
  int idx = blockIdx.x * 256 + threadIdx.x;
  if (idx >= 9 * 2 * 4 * 32 * 16) return;
  int j = idx & 15, lane = (idx >> 4) & 31, q = idx >> 9;
  int nt = q & 3; q >>= 2;
  int kt = q & 1; int tap = q >> 1;
  int n = nt * 16 + (lane & 15);
  int khalf = lane >> 4;
  int k = kt * 32 + ((j < 8) ? (khalf * 8 + j) : (16 + khalf * 8 + (j - 8)));
  dst[idx] = f2bf(src[(n * 64 + k) * 9 + tap]);
}

// ---------------------------------------------------------------------------
// Stage 0: bilinear-upsample sup -> conf, gate input, transpose to
// sequence-major bf16: xbuf[pix][t][c], confb[pix][t].
// One block per (b,t,h) row; 128 threads (w).
// ---------------------------------------------------------------------------

__global__ void __launch_bounds__(128)
x0_kernel(const float* __restrict__ tf, const float* __restrict__ sup,
          bf16_t* __restrict__ xbuf, float* __restrict__ confb) {
  __shared__ bf16_t vals[C_][W_];
  int h = blockIdx.x & 127;
  int t = (blockIdx.x >> 7) & 7;
  int b = blockIdx.x >> 10;
  int w = threadIdx.x;

  // align_corners=False bilinear, 32 -> 128 (scale 4)
  float sy = (h + 0.5f) * 0.25f - 0.5f;
  float sx = (w + 0.5f) * 0.25f - 0.5f;
  float fy0 = floorf(sy), fx0 = floorf(sx);
  float wy = fminf(fmaxf(sy - fy0, 0.0f), 1.0f);
  float wx = fminf(fmaxf(sx - fx0, 0.0f), 1.0f);
  int y0 = (int)fy0, x0 = (int)fx0;
  int y1 = y0 + 1, x1 = x0 + 1;
  y0 = y0 < 0 ? 0 : (y0 > 31 ? 31 : y0);
  y1 = y1 < 0 ? 0 : (y1 > 31 ? 31 : y1);
  x0 = x0 < 0 ? 0 : (x0 > 31 ? 31 : x0);
  x1 = x1 < 0 ? 0 : (x1 > 31 ? 31 : x1);
  const float* sp = sup + (((long)(b * T_ + t)) << 10);
  float v00 = sp[y0 * 32 + x0], v01 = sp[y0 * 32 + x1];
  float v10 = sp[y1 * 32 + x0], v11 = sp[y1 * 32 + x1];
  float v = (1.0f - wy) * ((1.0f - wx) * v00 + wx * v01)
          +         wy  * ((1.0f - wx) * v10 + wx * v11);
  float conf = fminf(fmaxf(1.0f - v, 0.0f), 1.0f);

  long p0 = ((long)(b * H_ + h)) * W_;
  confb[(p0 + w) * T_ + t] = conf;

  const float* tfp = tf + ((((long)(b * T_ + t)) * C_) << 14) + (h << 7);
  for (int c = 0; c < C_; ++c)
    vals[c][w] = f2bf(tfp[((long)c << 14) + w] * conf);
  __syncthreads();

  // coalesced 16B writes of [t][c] slices per pixel
  for (int it = 0; it < 8; ++it) {
    int chunk = it * 128 + threadIdx.x;
    int pix = chunk >> 3, part = chunk & 7;
    v8bf tmp;
    #pragma unroll
    for (int j = 0; j < 8; ++j) tmp[j] = vals[part * 8 + j][pix];
    *(v8bf*)&xbuf[((p0 + pix) * T_ + t) * C_ + part * 8] = tmp;
  }
}

// ---------------------------------------------------------------------------
// Fused SSM block: per wave 2 pixels => M=16 rows (2 pix x 8 timesteps).
//   h = x@Win;  h += gelu(h@W1+b1)@W2+b2;  g = sig(h@Wbg+bbg)*sig(conf*wsg+bsg)
//   s_t = (1-g)s + g h  (scan == serial loop over D-frag components == t)
//   out = s@Wout  ->  next x, or time-mean -> y (NHWC bf16) on last block.
// Swizzled weights staged once per workgroup in LDS.
// ---------------------------------------------------------------------------

#define SSM_OFF_IN   0
#define SSM_OFF_F1   4096
#define SSM_OFF_F2   12288
#define SSM_OFF_BG   20480
#define SSM_OFF_OUT  24576
#define SSM_W_ELEMS  28672
#define SSM_TILES_OFF 57344                       // bytes: weights region size
#define SSM_LDS_BYTES (57344 + 8*8192 + 8*16*4)   // + per-wave tiles + conf

__global__ void __launch_bounds__(256)
ssm_kernel(const bf16_t* __restrict__ xin, const float* __restrict__ confb,
           const bf16_t* __restrict__ sw,
           const float* __restrict__ b1, const float* __restrict__ b2,
           const float* __restrict__ bbg, const float* __restrict__ wsg,
           const float* __restrict__ bsg,
           bf16_t* __restrict__ xout, bf16_t* __restrict__ ynhwc, int last) {
  extern __shared__ char smem[];
  bf16_t* lw = (bf16_t*)smem;
  int tid = threadIdx.x;
  {  // stage swizzled weights: 57344B as dwords
    const int* s = (const int*)sw;
    int* d = (int*)lw;
    for (int i = tid; i < SSM_W_ELEMS / 2; i += 256) d[i] = s[i];
  }
  int wave = tid >> 5, lane = tid & 31;
  bf16_t* xt = (bf16_t*)(smem + SSM_TILES_OFF + wave * 8192);  // 16x64
  bf16_t* ht = xt + 1024;                                      // 16x64
  bf16_t* ut = ht + 1024;                                      // 16x128
  float* cfw = (float*)(smem + SSM_TILES_OFF + 8 * 8192) + wave * 16;

  long pix0 = ((long)blockIdx.x * 8 + wave) * 2;
  {  // stage x tile (2 pixels * 8t * 64c bf16 = 512 dwords)
    const int* s = (const int*)(xin + pix0 * (T_ * C_));
    int* d = (int*)xt;
    for (int i = lane; i < 512; i += 32) d[i] = s[i];
    if (lane < 16) cfw[lane] = confb[pix0 * T_ + lane];
  }
  __syncthreads();

  // ---- in_proj: h = x @ Win --------------------------------------------
  v8f hacc[4];
  {
    v16bf a0 = load_a(xt, 64, 0, lane);
    v16bf a1 = load_a(xt, 64, 32, lane);
    #pragma unroll
    for (int nt = 0; nt < 4; ++nt) {
      v8f acc = {};
      acc = wmma_bf16(a0, load16(lw + SSM_OFF_IN + ((0 * 4 + nt) * 32 + lane) * 16), acc);
      acc = wmma_bf16(a1, load16(lw + SSM_OFF_IN + ((1 * 4 + nt) * 32 + lane) * 16), acc);
      hacc[nt] = acc;
      int col = nt * 16 + (lane & 15), rb = (lane >> 4) * 8;
      #pragma unroll
      for (int v = 0; v < 8; ++v) ht[(rb + v) * 64 + col] = f2bf(acc[v]);
    }
  }
  WAVE_SYNC();

  // ---- ffn1: u = gelu(h @ W1 + b1)  (16 x 128) --------------------------
  {
    v16bf a0 = load_a(ht, 64, 0, lane);
    v16bf a1 = load_a(ht, 64, 32, lane);
    #pragma unroll
    for (int nt = 0; nt < 8; ++nt) {
      v8f acc = {};
      acc = wmma_bf16(a0, load16(lw + SSM_OFF_F1 + ((0 * 8 + nt) * 32 + lane) * 16), acc);
      acc = wmma_bf16(a1, load16(lw + SSM_OFF_F1 + ((1 * 8 + nt) * 32 + lane) * 16), acc);
      int col = nt * 16 + (lane & 15), rb = (lane >> 4) * 8;
      float bias = b1[col];
      #pragma unroll
      for (int v = 0; v < 8; ++v)
        ut[(rb + v) * 128 + col] = f2bf(gelu_f(acc[v] + bias));
    }
  }
  WAVE_SYNC();

  // ---- ffn2 + residual: h2 = h + u @ W2 + b2 ; stash into xt ------------
  {
    v16bf au0 = load_a(ut, 128, 0, lane);
    v16bf au1 = load_a(ut, 128, 32, lane);
    v16bf au2 = load_a(ut, 128, 64, lane);
    v16bf au3 = load_a(ut, 128, 96, lane);
    #pragma unroll
    for (int nt = 0; nt < 4; ++nt) {
      v8f acc = {};
      acc = wmma_bf16(au0, load16(lw + SSM_OFF_F2 + ((0 * 4 + nt) * 32 + lane) * 16), acc);
      acc = wmma_bf16(au1, load16(lw + SSM_OFF_F2 + ((1 * 4 + nt) * 32 + lane) * 16), acc);
      acc = wmma_bf16(au2, load16(lw + SSM_OFF_F2 + ((2 * 4 + nt) * 32 + lane) * 16), acc);
      acc = wmma_bf16(au3, load16(lw + SSM_OFF_F2 + ((3 * 4 + nt) * 32 + lane) * 16), acc);
      int col = nt * 16 + (lane & 15), rb = (lane >> 4) * 8;
      float bias = b2[col];
      #pragma unroll
      for (int v = 0; v < 8; ++v) {
        float h2 = hacc[nt][v] + acc[v] + bias;
        hacc[nt][v] = h2;
        xt[(rb + v) * 64 + col] = f2bf(h2);
      }
    }
  }
  WAVE_SYNC();

  // ---- gates + affine recurrence; s written into ht ---------------------
  {
    v16bf a0 = load_a(xt, 64, 0, lane);
    v16bf a1 = load_a(xt, 64, 32, lane);
    float cf[8];
    #pragma unroll
    for (int v = 0; v < 8; ++v) cf[v] = cfw[(lane >> 4) * 8 + v];
    #pragma unroll
    for (int nt = 0; nt < 4; ++nt) {
      v8f acc = {};
      acc = wmma_bf16(a0, load16(lw + SSM_OFF_BG + ((0 * 4 + nt) * 32 + lane) * 16), acc);
      acc = wmma_bf16(a1, load16(lw + SSM_OFF_BG + ((1 * 4 + nt) * 32 + lane) * 16), acc);
      int col = nt * 16 + (lane & 15), rb = (lane >> 4) * 8;
      float bg = bbg[col], wsv = wsg[col], bsv = bsg[col];
      float s = 0.0f;
      #pragma unroll
      for (int v = 0; v < 8; ++v) {   // v == timestep within this pixel
        float g = sigm_f(acc[v] + bg) * sigm_f(cf[v] * wsv + bsv);
        s = (1.0f - g) * s + g * hacc[nt][v];
        ht[(rb + v) * 64 + col] = f2bf(s);
      }
    }
  }
  WAVE_SYNC();

  // ---- out_proj ---------------------------------------------------------
  {
    v16bf a0 = load_a(ht, 64, 0, lane);
    v16bf a1 = load_a(ht, 64, 32, lane);
    #pragma unroll
    for (int nt = 0; nt < 4; ++nt) {
      v8f acc = {};
      acc = wmma_bf16(a0, load16(lw + SSM_OFF_OUT + ((0 * 4 + nt) * 32 + lane) * 16), acc);
      acc = wmma_bf16(a1, load16(lw + SSM_OFF_OUT + ((1 * 4 + nt) * 32 + lane) * 16), acc);
      int col = nt * 16 + (lane & 15), rb = (lane >> 4) * 8;
      if (!last) {
        #pragma unroll
        for (int v = 0; v < 8; ++v) {
          int row = rb + v;
          long pix = pix0 + (row >> 3);
          int t = row & 7;
          xout[(pix * T_ + t) * C_ + col] = f2bf(acc[v]);
        }
      } else {
        float m = 0.0f;
        #pragma unroll
        for (int v = 0; v < 8; ++v) m += acc[v];
        long pix = pix0 + (lane >> 4);
        ynhwc[pix * C_ + col] = f2bf(m * 0.125f);
      }
    }
  }
}

// ---------------------------------------------------------------------------
// Conv 3x3 (SAME) 64->64 as implicit GEMM: per wave one strip of 16 pixels,
// K = 9 taps x 64 ch = 18 WMMA k-steps. Deterministic GN partial sums.
// ---------------------------------------------------------------------------

#define CONV_W_ELEMS (9 * 2 * 4 * 32 * 16)  // 36864 bf16
#define CONV_LDS_BYTES (CONV_W_ELEMS * 2 + 2 * 8 * 256 * 4)

__global__ void __launch_bounds__(256)
conv3x3_kernel(const bf16_t* __restrict__ in_nhwc, const bf16_t* __restrict__ cw,
               float* __restrict__ zout, float* __restrict__ part) {
  extern __shared__ char smem[];
  bf16_t* lw = (bf16_t*)smem;
  float* redsum = (float*)(smem + CONV_W_ELEMS * 2);  // [8][256]
  float* redsq = redsum + 8 * 256;
  int tid = threadIdx.x;
  {
    const int* s = (const int*)cw;
    int* d = (int*)lw;
    for (int i = tid; i < CONV_W_ELEMS / 2; i += 256) d[i] = s[i];
  }
  #pragma unroll
  for (int g = 0; g < 8; ++g) { redsum[g * 256 + tid] = 0.0f; redsq[g * 256 + tid] = 0.0f; }
  __syncthreads();

  int wave = tid >> 5, lane = tid & 31;
  int sid = blockIdx.x * 8 + wave;  // strip id: b*1024 + h*8 + (w/16)
  int b = sid >> 10;
  int rem = sid & 1023;
  int h = rem >> 3;
  int w0 = (rem & 7) * 16;
  int prow = lane & 15;
  int khalf = lane >> 4;

  v8f zf = {};
  v8f acc[4] = {zf, zf, zf, zf};

  #pragma unroll
  for (int kh = 0; kh < 3; ++kh) {
    int ih = h + kh - 1;
    bool vh = (ih >= 0) && (ih < H_);
    #pragma unroll
    for (int kw = 0; kw < 3; ++kw) {
      int tap = kh * 3 + kw;
      int iw = w0 + prow + kw - 1;
      bool ok = vh && (iw >= 0) && (iw < W_);
      long base = (((long)(b * H_ + (ok ? ih : 0))) * W_ + (ok ? iw : 0)) * C_;
      #pragma unroll
      for (int kt = 0; kt < 2; ++kt) {
        v16bf a;
        if (ok) {
          const bf16_t* p = in_nhwc + base + kt * 32 + khalf * 8;
          v8bf lo = *(const v8bf*)p;
          v8bf hi = *(const v8bf*)(p + 16);
          a = __builtin_shufflevector(lo, hi, 0,1,2,3,4,5,6,7,8,9,10,11,12,13,14,15);
        } else {
          v8bf zb = {};
          a = __builtin_shufflevector(zb, zb, 0,1,2,3,4,5,6,7,8,9,10,11,12,13,14,15);
        }
        #pragma unroll
        for (int nt = 0; nt < 4; ++nt)
          acc[nt] = wmma_bf16(a, load16(lw + ((tap * 2 + kt) * 4 + nt) * 512 + lane * 16),
                              acc[nt]);
      }
    }
  }

  // write z (NCHW f32) + per-thread GN partials (no atomics -> deterministic)
  int rb = khalf * 8;
  #pragma unroll
  for (int nt = 0; nt < 4; ++nt) {
    int c = nt * 16 + (lane & 15);
    float s = 0.0f, s2 = 0.0f;
    long obase = (((long)(b * C_ + c)) * H_ + h) * W_ + w0 + rb;
    #pragma unroll
    for (int v = 0; v < 8; ++v) {
      float x = acc[nt][v];
      zout[obase + v] = x;
      s += x;
      s2 += x * x;
    }
    int g = c >> 3;
    redsum[g * 256 + tid] += s;
    redsq[g * 256 + tid] += s2;
  }
  __syncthreads();
  if (tid < 8) {  // fixed-order in-block reduction
    float s = 0.0f, s2 = 0.0f;
    for (int i = 0; i < 256; ++i) { s += redsum[tid * 256 + i]; s2 += redsq[tid * 256 + i]; }
    part[blockIdx.x * 16 + tid * 2 + 0] = s;
    part[blockIdx.x * 16 + tid * 2 + 1] = s2;
  }
}

// fixed-order global reduce: stats[b*16 + g*2 + {sum,sumsq}]
__global__ void gn_reduce_kernel(const float* __restrict__ part, float* __restrict__ stats) {
  int tid = threadIdx.x;
  if (tid >= 32) return;
  int b = tid >> 4, j = tid & 15;
  float s = 0.0f;
  for (int k = 0; k < 128; ++k) s += part[((b * 128 + k) * 16) + j];
  stats[tid] = s;
}

// GN + GELU (+ residual add / NHWC re-layout depending on stage)
__global__ void __launch_bounds__(256)
gn_apply_kernel(const float* __restrict__ z, const float* __restrict__ stats,
                const float* __restrict__ gamma, const float* __restrict__ beta,
                bf16_t* __restrict__ out_nhwc, float* __restrict__ out_f32,
                const float* __restrict__ resid, int mode) {
  long idx = (long)blockIdx.x * 256 + threadIdx.x;  // over B*C*H*W (w fastest)
  int b = (int)(idx >> 20);
  int c = (int)((idx >> 14) & 63);
  int hw = (int)(idx & 16383);
  int g = c >> 3;
  const float inv_cnt = 1.0f / 131072.0f;           // 8 ch * 128 * 128
  float mu = stats[b * 16 + g * 2 + 0] * inv_cnt;
  float var = stats[b * 16 + g * 2 + 1] * inv_cnt - mu * mu;
  float rs = rsqrtf(var + 1e-5f);
  float val = gelu_f((z[idx] - mu) * rs * gamma[c] + beta[c]);
  if (mode == 2) {
    out_f32[idx] = resid[idx] + val;                // final: y1 + r2 -> d_out
  } else {
    out_nhwc[(((long)b * 16384 + hw) * 64) + c] = f2bf(val);
    if (mode == 0) out_f32[idx] = val;              // keep y1 f32 for residual
  }
}

// ---------------------------------------------------------------------------
// Host orchestration
// ---------------------------------------------------------------------------

extern "C" void kernel_launch(void* const* d_in, const int* in_sizes, int n_in,
                              void* d_out, int out_size, void* d_ws, size_t ws_size,
                              hipStream_t stream) {
  (void)in_sizes; (void)n_in; (void)out_size; (void)ws_size;
  const float* tf = (const float*)d_in[0];
  const float* sup = (const float*)d_in[1];
  // params flatten order: per block {in.w, out.w, bg.w, bg.b, sg.w, sg.b,
  //  f1.w, f1.b, f2.w, f2.b}, then post0/res0/res1 {conv, gn_w, gn_b}.
  char* ws = (char*)d_ws;
  size_t cur = 0;
  auto alloc = [&](size_t bytes) {
    cur = (cur + 255) & ~(size_t)255;
    size_t o = cur;
    cur += bytes;
    return o;
  };
  size_t off_sw0 = alloc(SSM_W_ELEMS * 2);
  size_t off_sw1 = alloc(SSM_W_ELEMS * 2);
  size_t off_cw0 = alloc(CONV_W_ELEMS * 2);
  size_t off_cw1 = alloc(CONV_W_ELEMS * 2);
  size_t off_cw2 = alloc(CONV_W_ELEMS * 2);
  size_t off_part = alloc(256 * 16 * 4);
  size_t off_stats = alloc(32 * 4);
  size_t off_conf = alloc((size_t)NPIX * T_ * 4);
  size_t off_xA = alloc((size_t)NPIX * T_ * C_ * 2);   // 32 MB, later reused
  size_t off_xB = alloc((size_t)NPIX * T_ * C_ * 2);   // 32 MB
  // aliases inside xA (dead after ssm block 0 consumed it):
  const size_t MB = 1024u * 1024u;
  size_t off_ynhwc  = off_xA;            // 4 MB
  size_t off_z      = off_xA + 4 * MB;   // 8 MB
  size_t off_y1nhwc = off_xA + 12 * MB;  // 4 MB
  size_t off_y1f32  = off_xA + 16 * MB;  // 8 MB
  size_t off_r1nhwc = off_xA + 24 * MB;  // 4 MB

  // --- weight swizzle -----------------------------------------------------
  for (int blk = 0; blk < 2; ++blk) {
    int base = 2 + blk * 10;
    bf16_t* sw = (bf16_t*)(ws + (blk ? off_sw1 : off_sw0));
    swz_linear_kernel<<<16, 256, 0, stream>>>((const float*)d_in[base + 0], sw + SSM_OFF_IN, 64, 64);
    swz_linear_kernel<<<32, 256, 0, stream>>>((const float*)d_in[base + 6], sw + SSM_OFF_F1, 64, 128);
    swz_linear_kernel<<<32, 256, 0, stream>>>((const float*)d_in[base + 8], sw + SSM_OFF_F2, 128, 64);
    swz_linear_kernel<<<16, 256, 0, stream>>>((const float*)d_in[base + 2], sw + SSM_OFF_BG, 64, 64);
    swz_linear_kernel<<<16, 256, 0, stream>>>((const float*)d_in[base + 1], sw + SSM_OFF_OUT, 64, 64);
  }
  swz_conv_kernel<<<144, 256, 0, stream>>>((const float*)d_in[22], (bf16_t*)(ws + off_cw0));
  swz_conv_kernel<<<144, 256, 0, stream>>>((const float*)d_in[25], (bf16_t*)(ws + off_cw1));
  swz_conv_kernel<<<144, 256, 0, stream>>>((const float*)d_in[28], (bf16_t*)(ws + off_cw2));

  // --- gate + layout ------------------------------------------------------
  x0_kernel<<<B_ * T_ * H_, 128, 0, stream>>>(tf, sup, (bf16_t*)(ws + off_xA),
                                              (float*)(ws + off_conf));

  // --- two fused SSM blocks ----------------------------------------------
  for (int blk = 0; blk < 2; ++blk) {
    int base = 2 + blk * 10;
    ssm_kernel<<<NPIX / 16, 256, SSM_LDS_BYTES, stream>>>(
        (const bf16_t*)(ws + (blk ? off_xB : off_xA)),
        (const float*)(ws + off_conf),
        (const bf16_t*)(ws + (blk ? off_sw1 : off_sw0)),
        (const float*)d_in[base + 7],   // ffn1.b
        (const float*)d_in[base + 9],   // ffn2.b
        (const float*)d_in[base + 3],   // base_gate.b
        (const float*)d_in[base + 4],   // sup_gate.w
        (const float*)d_in[base + 5],   // sup_gate.b
        (bf16_t*)(ws + off_xB),
        (bf16_t*)(ws + off_ynhwc),
        blk == 1);
  }

  // --- conv / GN / GELU stages -------------------------------------------
  // stage 0: post0(y) -> y1 (bf16 NHWC + f32 NCHW)
  conv3x3_kernel<<<256, 256, CONV_LDS_BYTES, stream>>>(
      (const bf16_t*)(ws + off_ynhwc), (const bf16_t*)(ws + off_cw0),
      (float*)(ws + off_z), (float*)(ws + off_part));
  gn_reduce_kernel<<<1, 32, 0, stream>>>((const float*)(ws + off_part), (float*)(ws + off_stats));
  gn_apply_kernel<<<8192, 256, 0, stream>>>(
      (const float*)(ws + off_z), (const float*)(ws + off_stats),
      (const float*)d_in[23], (const float*)d_in[24],
      (bf16_t*)(ws + off_y1nhwc), (float*)(ws + off_y1f32), nullptr, 0);

  // stage 1: res0(y1) -> r1 (bf16 NHWC)
  conv3x3_kernel<<<256, 256, CONV_LDS_BYTES, stream>>>(
      (const bf16_t*)(ws + off_y1nhwc), (const bf16_t*)(ws + off_cw1),
      (float*)(ws + off_z), (float*)(ws + off_part));
  gn_reduce_kernel<<<1, 32, 0, stream>>>((const float*)(ws + off_part), (float*)(ws + off_stats));
  gn_apply_kernel<<<8192, 256, 0, stream>>>(
      (const float*)(ws + off_z), (const float*)(ws + off_stats),
      (const float*)d_in[26], (const float*)d_in[27],
      (bf16_t*)(ws + off_r1nhwc), nullptr, nullptr, 1);

  // stage 2: res1(r1) -> r2; out = y1 + r2 (fp32 NCHW -> d_out)
  conv3x3_kernel<<<256, 256, CONV_LDS_BYTES, stream>>>(
      (const bf16_t*)(ws + off_r1nhwc), (const bf16_t*)(ws + off_cw2),
      (float*)(ws + off_z), (float*)(ws + off_part));
  gn_reduce_kernel<<<1, 32, 0, stream>>>((const float*)(ws + off_part), (float*)(ws + off_stats));
  gn_apply_kernel<<<8192, 256, 0, stream>>>(
      (const float*)(ws + off_z), (const float*)(ws + off_stats),
      (const float*)d_in[29], (const float*)d_in[30],
      nullptr, (float*)d_out, (const float*)(ws + off_y1f32), 2);
}
